// MambaDecoderLayer_34437047779934
// MI455X (gfx1250) — matrix operand order repaired and checked by reading, not compiled, not run on hardware
//
#include <hip/hip_runtime.h>
#include <hip/hip_bf16.h>
#include <math.h>

// ---------------- model dims ----------------
#define LSEQ     2048
#define DMODEL   2048
#define DINNER   2048
#define DXB      512
#define NHEADS   32
#define HEADDIM  64
#define DSTATE   64
#define NKV      8
#define REP      4
#define DFF      8192
#define DCONV    4
#define DPROJ    (2*DINNER + 2*DXB + NHEADS)   // 5152
#define CONVCH   (2*DXB + DINNER)              // 3072
#define EPSV     1e-5f

typedef __bf16 v16bf __attribute__((ext_vector_type(16)));
typedef float  v8f   __attribute__((ext_vector_type(8)));
typedef unsigned int u32x4 __attribute__((ext_vector_type(4)));
typedef int          i32x4 __attribute__((ext_vector_type(4)));
typedef int          i32x8 __attribute__((ext_vector_type(8)));

union FragU { uint4 q[2]; v16bf v; };

static __device__ __forceinline__ unsigned short f2bf(float f) {
    unsigned int u = __float_as_uint(f);
    u = u + 0x7FFFu + ((u >> 16) & 1u);   // round-to-nearest-even
    return (unsigned short)(u >> 16);
}
static __device__ __forceinline__ float siluf(float x) {
    return x / (1.f + expf(-x));
}

// ---- TDM: async 2D bf16 tile load global->LDS, padded rows (64 elem + 16 pad = 80) ----
// D# per CDNA5 ISA: group0 {count,lds_addr,global_addr,type=2},
// group1 {data_size=2B, pad_enable, pad_interval=4 (128B), pad_amount=7 (+32B),
//         tensor_dim0=K, tensor_dim1=M, tile_dim0=64, tile_dim1=rows, dim0_stride=K}
static __device__ __forceinline__ void tdm_load_tile_bf16(
    unsigned lds_addr, const unsigned short* gsrc, int Klen, int Mlen, int tileRows)
{
    unsigned long long ga = (unsigned long long)(const void*)gsrc;
    u32x4 g0;
    g0[0] = 1u;                                    // count=1 (valid user descriptor)
    g0[1] = lds_addr;                              // LDS byte address
    g0[2] = (unsigned)(ga & 0xFFFFFFFFu);
    g0[3] = (unsigned)((ga >> 32) & 0x01FFFFFFu) | (2u << 30);  // addr[56:32], type=2
    i32x8 g1;
    g1[0] = (1 << 16) | (1 << 20) | (4 << 22) | (7 << 25);      // size=2B, pad on
    g1[1] = (Klen & 0xFFFF) << 16;                              // tensor_dim0 lo16
    g1[2] = ((Klen >> 16) & 0xFFFF) | ((Mlen & 0xFFFF) << 16);  // dim0 hi / dim1 lo
    g1[3] = ((Mlen >> 16) & 0xFFFF) | (64 << 16);               // dim1 hi / tile_dim0=64
    g1[4] = (tileRows & 0xFFFF);                                // tile_dim1, tile_dim2=0
    g1[5] = Klen;                                               // tensor_dim0_stride lo32
    g1[6] = 0;
    g1[7] = 0;
    i32x4 z4 = {0, 0, 0, 0};
#if __clang_major__ >= 23
    i32x8 z8 = {0, 0, 0, 0, 0, 0, 0, 0};
    __builtin_amdgcn_tensor_load_to_lds(g0, g1, z4, z4, z8, 0);
#else
    __builtin_amdgcn_tensor_load_to_lds(g0, g1, z4, z4, 0);
#endif
}

// ---------------- fp32 -> bf16 convert ----------------
__global__ void __launch_bounds__(256) cvt_bf16(const float* __restrict__ in,
                                                unsigned short* __restrict__ out, int n) {
    int i = blockIdx.x * 256 + threadIdx.x;
    if (i < n) out[i] = f2bf(in[i]);
}

// ---------------- RMSNorm (f32 in, bf16 out), one block per row ----------------
__global__ void __launch_bounds__(256) rmsnorm_bf16(const float* __restrict__ x,
                                                    const float* __restrict__ w,
                                                    unsigned short* __restrict__ out, int D) {
    int row = blockIdx.x;
    const float* xr = x + (size_t)row * D;
    float s = 0.f;
    for (int c = threadIdx.x; c < D; c += 256) { float v = xr[c]; s += v * v; }
    __shared__ float red[256];
    red[threadIdx.x] = s; __syncthreads();
    for (int st = 128; st > 0; st >>= 1) {
        if (threadIdx.x < st) red[threadIdx.x] += red[threadIdx.x + st];
        __syncthreads();
    }
    float scale = rsqrtf(red[0] / (float)D + EPSV);
    for (int c = threadIdx.x; c < D; c += 256)
        out[(size_t)row * D + c] = f2bf(xr[c] * scale * w[c]);
}

// ---------------- gated RMSNorm: v = y*silu(z); rmsnorm(v)*w -> bf16 ----------------
__global__ void __launch_bounds__(256) gated_rmsnorm_bf16(const float* __restrict__ y,
                                                          const float* __restrict__ zx,
                                                          const float* __restrict__ w,
                                                          unsigned short* __restrict__ out) {
    int row = blockIdx.x;
    float vals[DINNER / 256];
    float s = 0.f;
    #pragma unroll
    for (int i = 0; i < DINNER / 256; i++) {
        int c = threadIdx.x + i * 256;
        float z = zx[(size_t)row * DPROJ + c];
        float v = y[(size_t)row * DINNER + c] * siluf(z);
        vals[i] = v; s += v * v;
    }
    __shared__ float red[256];
    red[threadIdx.x] = s; __syncthreads();
    for (int st = 128; st > 0; st >>= 1) {
        if (threadIdx.x < st) red[threadIdx.x] += red[threadIdx.x + st];
        __syncthreads();
    }
    float scale = rsqrtf(red[0] / (float)DINNER + EPSV);
    #pragma unroll
    for (int i = 0; i < DINNER / 256; i++) {
        int c = threadIdx.x + i * 256;
        out[(size_t)row * DINNER + c] = f2bf(vals[i] * scale * w[c]);
    }
}

// ---------------- causal depthwise conv + SiLU over concat(x,B,C) ----------------
__global__ void __launch_bounds__(256) conv_silu(const float* __restrict__ zx,
                                                 const float* __restrict__ cw,
                                                 const float* __restrict__ cb,
                                                 float* __restrict__ xBC) {
    int i = blockIdx.x * 256 + threadIdx.x;
    if (i >= LSEQ * CONVCH) return;
    int t = i / CONVCH, c = i % CONVCH;
    float y = cb[c];
    #pragma unroll
    for (int k = 0; k < DCONV; k++) {
        int tt = t - (DCONV - 1) + k;
        if (tt >= 0) y += zx[(size_t)tt * DPROJ + DINNER + c] * cw[c * DCONV + k];
    }
    xBC[i] = siluf(y);
}

// ---------------- dt = softplus(raw + bias), dA = exp(-exp(A_log)*dt) ----------------
__global__ void __launch_bounds__(256) dt_prep(const float* __restrict__ zx,
                                               const float* __restrict__ dt_bias,
                                               const float* __restrict__ A_log,
                                               float* __restrict__ dtv, float* __restrict__ dAv) {
    int i = blockIdx.x * 256 + threadIdx.x;
    if (i >= LSEQ * NHEADS) return;
    int t = i >> 5, h = i & 31;
    float v = zx[(size_t)t * DPROJ + 2 * DINNER + 2 * DXB + h] + dt_bias[h];
    float dt = (v > 20.f) ? v : log1pf(expf(v));
    dtv[i] = dt;
    dAv[i] = expf(-expf(A_log[h]) * dt);
}

// ---------------- selective scan: one block per head, state in registers ----------------
__global__ void __launch_bounds__(256) selective_scan(const float* __restrict__ xBC,
                                                      const float* __restrict__ dtv,
                                                      const float* __restrict__ dAv,
                                                      const float* __restrict__ Dp,
                                                      float* __restrict__ y) {
    int h = blockIdx.x;           // 0..31
    int kv = h >> 2;              // GQA: REP=4
    int tid = threadIdx.x;
    int p = tid >> 2;             // 0..63
    int nb = (tid & 3) << 4;      // 16-wide n slice
    const float Dh = Dp[h];
    __shared__ float sx[HEADDIM], sB[DSTATE], sC[DSTATE], sScal[2];
    float hreg[16];
    #pragma unroll
    for (int i = 0; i < 16; i++) hreg[i] = 0.f;

    for (int t = 0; t < LSEQ; t++) {
        const float* row = xBC + (size_t)t * CONVCH;
        if (tid < 64)        sx[tid]        = row[kv * 64 + tid];
        else if (tid < 128)  sB[tid - 64]   = row[DXB + kv * 64 + (tid - 64)];
        else if (tid < 192)  sC[tid - 128]  = row[2 * DXB + h * 64 + (tid - 128)];
        else if (tid == 192) { sScal[0] = dtv[t * NHEADS + h]; sScal[1] = dAv[t * NHEADS + h]; }
        __syncthreads();
        float xp = sx[p];
        float coef = sScal[0] * xp;
        float a = sScal[1];
        float sum = 0.f;
        #pragma unroll
        for (int i = 0; i < 16; i++) {
            hreg[i] = hreg[i] * a + coef * sB[nb + i];
            sum += hreg[i] * sC[nb + i];
        }
        sum += __shfl_xor(sum, 1, 32);   // wave32: 4 consecutive lanes share one p
        sum += __shfl_xor(sum, 2, 32);
        if ((tid & 3) == 0) y[(size_t)t * DINNER + h * HEADDIM + p] = sum + Dh * xp;
        __syncthreads();
    }
}

// ---------------- bf16 WMMA GEMM: C(MxN) = A(MxK) * W(NxK)^T ----------------
// Block tile 128x64, 8 waves of 32x32 (4 accumulators), BK=64.
// A tile arrives by TDM (double-buffered, padded stride 80); W tile staged row-major [n][k].
// EPI 0: outF = acc ; EPI 1: outF = acc + aux ; EPI 2: outB = bf16(silu(aux)*acc)
template <int EPI>
__global__ void __launch_bounds__(256) gemm_bf16w(const unsigned short* __restrict__ A,
                                                  const unsigned short* __restrict__ W,
                                                  int M, int N, int K,
                                                  const float* __restrict__ aux,
                                                  float* __restrict__ outF,
                                                  unsigned short* __restrict__ outB) {
    constexpr int LS = 80;                                  // padded LDS row stride (elems)
    __shared__ __align__(16) unsigned short lsA[2][128 * LS]; // A tiles [m][k], TDM-filled
    __shared__ __align__(16) unsigned short lsW[64 * LS];     // W tile [n][k], padded stride 80
    int tid  = threadIdx.x;
    int lane = tid & 31;
    int wave = tid >> 5;
    int wm = wave & 3;        // 4 M sub-tiles of 32
    int wn = wave >> 2;       // 2 N sub-tiles of 32
    int m0 = blockIdx.x * 128;
    int n0 = blockIdx.y * 64;
    v8f acc00 = {}, acc01 = {}, acc10 = {}, acc11 = {};
    int r   = tid >> 2;       // 0..63
    int c16 = (tid & 3) << 4;
    const int nK = K >> 6;
    const bool tdmWave = (wave == 0);

    if (tdmWave)
        tdm_load_tile_bf16((unsigned)(size_t)&lsA[0][0],
                           A + (size_t)m0 * K, K, M, 128);

    for (int kt = 0; kt < nK; kt++) {
        int k0 = kt << 6;
        int cur = kt & 1;
        // stage W tile into [n][k] (all waves)
        {
            int nr = n0 + r; if (nr > N - 1) nr = N - 1;
            const unsigned short* g = W + (size_t)nr * K + k0 + c16;
            #pragma unroll
            for (int j = 0; j < 16; j++) lsW[r * LS + c16 + j] = g[j];
            if (kt + 1 < nK)
                __builtin_prefetch(W + (size_t)nr * K + k0 + 64 + c16, 0, 1);
        }
        // wave0: kick next A tile DMA, then wait for current tile to land
        if (tdmWave) {
            if (kt + 1 < nK) {
                tdm_load_tile_bf16((unsigned)(size_t)&lsA[cur ^ 1][0],
                                   A + (size_t)m0 * K + (k0 + 64), K, M, 128);
                __builtin_amdgcn_s_wait_tensorcnt(1);   // TDM in-order: current done
            } else {
                __builtin_amdgcn_s_wait_tensorcnt(0);
            }
        }
        __syncthreads();
        const unsigned short* a_base = &lsA[cur][0];
        #pragma unroll
        for (int ks = 0; ks < 2; ks++) {
            int kk = ks << 5;
            FragU a0, a1, b0, b1;
            // A 16x32 bf16 fragment: lane holds row m, K split per ISA layout
            int am = wm * 32 + (lane & 15);
            int ak = kk + ((lane >> 4) << 3);
            a0.q[0] = *(const uint4*)&a_base[am * LS + ak];
            a0.q[1] = *(const uint4*)&a_base[am * LS + ak + 16];
            a1.q[0] = *(const uint4*)&a_base[(am + 16) * LS + ak];
            a1.q[1] = *(const uint4*)&a_base[(am + 16) * LS + ak + 16];
            // B 32x16 bf16 fragment: lane holds column n, K split per ISA layout
            int bk = kk + ((lane >> 4) << 3);
            int bn = wn * 32 + (lane & 15);
            b0.q[0] = *(const uint4*)&lsW[bn * LS + bk];
            b0.q[1] = *(const uint4*)&lsW[bn * LS + bk + 16];
            b1.q[0] = *(const uint4*)&lsW[(bn + 16) * LS + bk];
            b1.q[1] = *(const uint4*)&lsW[(bn + 16) * LS + bk + 16];
            acc00 = __builtin_amdgcn_wmma_f32_16x16x32_bf16(false, a0.v, false, b0.v,
                                                            (short)0, acc00, false, false);
            acc01 = __builtin_amdgcn_wmma_f32_16x16x32_bf16(false, a0.v, false, b1.v,
                                                            (short)0, acc01, false, false);
            acc10 = __builtin_amdgcn_wmma_f32_16x16x32_bf16(false, a1.v, false, b0.v,
                                                            (short)0, acc10, false, false);
            acc11 = __builtin_amdgcn_wmma_f32_16x16x32_bf16(false, a1.v, false, b1.v,
                                                            (short)0, acc11, false, false);
        }
        __syncthreads();
    }

    // C/D layout: VGPR rr -> M = rr (lanes 0-15) / rr+8 (lanes 16-31); N = lane&15
    int rowBase = m0 + wm * 32 + ((lane >> 4) << 3);
    int colBase = n0 + wn * 32 + (lane & 15);
    #pragma unroll
    for (int rr = 0; rr < 8; rr++) {
        #pragma unroll
        for (int mi = 0; mi < 2; mi++) {
            int row = rowBase + mi * 16 + rr;
            float v0 = (mi == 0) ? acc00[rr] : acc10[rr];
            float v1 = (mi == 0) ? acc01[rr] : acc11[rr];
            int c0 = colBase, c1 = colBase + 16;
            size_t i0 = (size_t)row * N + c0, i1 = (size_t)row * N + c1;
            if constexpr (EPI == 0) {
                if (c0 < N) outF[i0] = v0;
                if (c1 < N) outF[i1] = v1;
            } else if constexpr (EPI == 1) {
                if (c0 < N) outF[i0] = v0 + aux[i0];
                if (c1 < N) outF[i1] = v1 + aux[i1];
            } else {
                if (c0 < N) outB[i0] = f2bf(siluf(aux[i0]) * v0);
                if (c1 < N) outB[i1] = f2bf(siluf(aux[i1]) * v1);
            }
        }
    }
}

// ---------------- host launcher ----------------
extern "C" void kernel_launch(void* const* d_in, const int* in_sizes, int n_in,
                              void* d_out, int out_size, void* d_ws, size_t ws_size,
                              hipStream_t stream) {
    const float* hidden     = (const float*)d_in[0];
    const float* in_proj_w  = (const float*)d_in[1];
    const float* conv_w     = (const float*)d_in[2];
    const float* conv_b     = (const float*)d_in[3];
    const float* dt_bias    = (const float*)d_in[4];
    const float* A_log      = (const float*)d_in[5];
    const float* Dvec       = (const float*)d_in[6];
    const float* ssm_norm_w = (const float*)d_in[7];
    const float* out_proj_w = (const float*)d_in[8];
    const float* input_ln_w = (const float*)d_in[9];
    const float* post_ln_w  = (const float*)d_in[10];
    const float* gate_w     = (const float*)d_in[11];
    const float* up_w       = (const float*)d_in[12];
    const float* down_w     = (const float*)d_in[13];

    char* ws = (char*)d_ws;
    size_t off = 0;
    auto take = [&](size_t bytes) -> size_t {
        size_t o = off; off += (bytes + 255) & ~(size_t)255; return o;
    };
    unsigned short* hN1  = (unsigned short*)(ws + take((size_t)LSEQ * DMODEL * 2));
    float*          zx   = (float*)(ws + take((size_t)LSEQ * DPROJ * 4));
    float*          xBC  = (float*)(ws + take((size_t)LSEQ * CONVCH * 4));
    float*          dtv  = (float*)(ws + take((size_t)LSEQ * NHEADS * 4));
    float*          dAv  = (float*)(ws + take((size_t)LSEQ * NHEADS * 4));
    float*          yAcc = (float*)(ws + take((size_t)LSEQ * DINNER * 4));
    unsigned short* yG   = (unsigned short*)(ws + take((size_t)LSEQ * DINNER * 2));
    float*          h2   = (float*)(ws + take((size_t)LSEQ * DMODEL * 4));
    unsigned short* hN2  = (unsigned short*)(ws + take((size_t)LSEQ * DMODEL * 2));
    float*          gbuf = (float*)(ws + take((size_t)LSEQ * DFF * 4));
    unsigned short* ffb  = (unsigned short*)(ws + take((size_t)LSEQ * DFF * 2));
    unsigned short* Win  = (unsigned short*)(ws + take((size_t)DPROJ * DMODEL * 2));
    unsigned short* Wout = (unsigned short*)(ws + take((size_t)DMODEL * DINNER * 2));
    unsigned short* Wg   = (unsigned short*)(ws + take((size_t)DFF * DMODEL * 2));
    unsigned short* Wu   = (unsigned short*)(ws + take((size_t)DFF * DMODEL * 2));
    unsigned short* Wd   = (unsigned short*)(ws + take((size_t)DMODEL * DFF * 2));

    auto blocks = [](long long n) { return (unsigned)((n + 255) / 256); };

    // 1) weight conversion to bf16
    cvt_bf16<<<blocks((long long)DPROJ * DMODEL), 256, 0, stream>>>(in_proj_w, Win, DPROJ * DMODEL);
    cvt_bf16<<<blocks((long long)DMODEL * DINNER), 256, 0, stream>>>(out_proj_w, Wout, DMODEL * DINNER);
    cvt_bf16<<<blocks((long long)DFF * DMODEL), 256, 0, stream>>>(gate_w, Wg, DFF * DMODEL);
    cvt_bf16<<<blocks((long long)DFF * DMODEL), 256, 0, stream>>>(up_w, Wu, DFF * DMODEL);
    cvt_bf16<<<blocks((long long)DMODEL * DFF), 256, 0, stream>>>(down_w, Wd, DMODEL * DFF);

    // 2) input RMSNorm -> bf16
    rmsnorm_bf16<<<LSEQ, 256, 0, stream>>>(hidden, input_ln_w, hN1, DMODEL);

    // 3) in_proj GEMM: zxbcdt = hN1 @ Win^T  (2048 x 5152)
    gemm_bf16w<0><<<dim3(LSEQ / 128, (DPROJ + 63) / 64), 256, 0, stream>>>(
        hN1, Win, LSEQ, DPROJ, DMODEL, nullptr, zx, nullptr);

    // 4) causal conv + silu, 5) dt/dA prep
    conv_silu<<<blocks((long long)LSEQ * CONVCH), 256, 0, stream>>>(zx, conv_w, conv_b, xBC);
    dt_prep<<<blocks((long long)LSEQ * NHEADS), 256, 0, stream>>>(zx, dt_bias, A_log, dtv, dAv);

    // 6) selective scan (sequential over L, one block per head)
    selective_scan<<<NHEADS, 256, 0, stream>>>(xBC, dtv, dAv, Dvec, yAcc);

    // 7) gated RMSNorm -> bf16
    gated_rmsnorm_bf16<<<LSEQ, 256, 0, stream>>>(yAcc, zx, ssm_norm_w, yG);

    // 8) out_proj GEMM + residual(hidden) -> h2
    gemm_bf16w<1><<<dim3(LSEQ / 128, DMODEL / 64), 256, 0, stream>>>(
        yG, Wout, LSEQ, DMODEL, DINNER, hidden, h2, nullptr);

    // 9) post RMSNorm -> bf16
    rmsnorm_bf16<<<LSEQ, 256, 0, stream>>>(h2, post_ln_w, hN2, DMODEL);

    // 10) gate GEMM (f32), 11) up GEMM with fused silu(gate)*up -> bf16
    gemm_bf16w<0><<<dim3(LSEQ / 128, DFF / 64), 256, 0, stream>>>(
        hN2, Wg, LSEQ, DFF, DMODEL, nullptr, gbuf, nullptr);
    gemm_bf16w<2><<<dim3(LSEQ / 128, DFF / 64), 256, 0, stream>>>(
        hN2, Wu, LSEQ, DFF, DMODEL, gbuf, nullptr, ffb);

    // 12) down GEMM + residual(h2) -> d_out
    gemm_bf16w<1><<<dim3(LSEQ / 128, DMODEL / 64), 256, 0, stream>>>(
        ffb, Wd, LSEQ, DMODEL, DFF, h2, (float*)d_out, nullptr);
}
